// LongAttentionBERT_84722524881029
// MI455X (gfx1250) — matrix operand order repaired and checked by reading, not compiled, not run on hardware
//
#include <hip/hip_runtime.h>
#include <hip/hip_bf16.h>
#include <stdint.h>

// ---------------------------------------------------------------------------
// BigBird-style block-sparse attention BERT layer for MI455X (gfx1250).
// Layers in the reference do NOT chain; only hidden_states[L-1] reaches the
// output, so we compute exactly one layer + LN + classifier head.
// All matmuls use v_wmma_f32_16x16x32_f16 (f16 in, f32 accumulate).
// Attention K/V tiles are staged into LDS with the Tensor Data Mover
// (tensor_load_to_lds + s_wait_tensorcnt), double-buffered to overlap DMA
// with WMMA compute. GEMM K-loops are software-pipelined and 2x unrolled so
// the two fragment register sets ping-pong by name (no rotation copies).
// ---------------------------------------------------------------------------

#define Ldim  13
#define Ddim  768
#define Hdim  12
#define HDdim 64
#define Tdim  1024
#define NBdim 16      // T / 64 key/query blocks
#define Cdim  16
#define NEGV  (-1.0e9f)
#define SCALE 0.125f  // 1/sqrt(64)

typedef __attribute__((ext_vector_type(16))) _Float16 v16h;
typedef __attribute__((ext_vector_type(8)))  _Float16 v8h;
typedef __attribute__((ext_vector_type(8)))  float    v8f;
typedef __attribute__((ext_vector_type(4)))  unsigned int u32x4;
typedef __attribute__((ext_vector_type(8)))  int          i32x8;
typedef __attribute__((ext_vector_type(4)))  int          i32x4;

#if defined(__has_builtin)
#if __has_builtin(__builtin_amdgcn_tensor_load_to_lds) && \
    __has_builtin(__builtin_amdgcn_s_wait_tensorcnt)
#define HAVE_TDM 1
#endif
#endif
#ifndef HAVE_TDM
#define HAVE_TDM 0
#endif

// ---- WMMA fragment helpers (wave32, 16x16x32 f16) -------------------------
// A fragment: 16x32 (MxK). Lanes 0-15 hold M=lane, lanes 16-31 M=lane-16.
// Per lane: halfs 0..15 -> K = kh..kh+15,
// kh = 0 (lanes 0-15) or 16 (lanes 16-31).  Two contiguous 16B loads.
__device__ __forceinline__ v16h load_frag_a(const _Float16* base, int ld) {
  const int lane = threadIdx.x & 31;
  const int m    = lane & 15;
  const int kh   = (lane & 16) ? 16 : 0;
  const _Float16* p = base + m * ld + kh;
  v8h lo = *(const v8h*)(p);
  v8h hi = *(const v8h*)(p + 8);
  v16h r;
#pragma unroll
  for (int i = 0; i < 8; ++i) { r[i] = lo[i]; r[i + 8] = hi[i]; }
  return r;
}

// B fragment: 32x16 (KxN), element (k,n) taken from base[n*ld + k]
// (source stored N-major with K contiguous). Lanes 0-15: N=lane, K=0..15;
// lanes 16-31: N=lane-16, K=16..31. One contiguous 32B load per lane.
__device__ __forceinline__ v16h load_frag_b(const _Float16* base, int ld) {
  const int lane = threadIdx.x & 31;
  const int n    = lane & 15;
  const int kh   = (lane & 16) ? 16 : 0;
  return *(const v16h*)(base + n * ld + kh);
}

__device__ __forceinline__ v8f wmma_f16(v16h a, v16h b, v8f c) {
  return __builtin_amdgcn_wmma_f32_16x16x32_f16(false, a, false, b,
                                                (short)0, c, false, false);
}

#if HAVE_TDM
// ---- Tensor Data Mover: 2-D tile load (rows x rowLen f16 elements) --------
// D# encoding per CDNA5 ISA:
//   group0: count=1 | lds_addr | global_addr[56:0] | type=2
//   group1: data_size=1(2B) | tensor_dim0/1 | tile_dim0/1 | dim0 stride
// Tracked by TENSORcnt; caller pairs with s_wait_tensorcnt.
__device__ __forceinline__ void tdm_load_tile_2d(unsigned ldsOff,
                                                 const _Float16* g,
                                                 unsigned rows, unsigned rowLen,
                                                 unsigned strideElems) {
  unsigned long long ga = (unsigned long long)(uintptr_t)g;
  u32x4 g0 = {};
  g0[0] = 1u;                                          // count=1, user D#
  g0[1] = ldsOff;                                      // LDS byte address
  g0[2] = (unsigned)ga;                                // global_addr[31:0]
  g0[3] = (unsigned)((ga >> 32) & 0x01FFFFFFull) | (2u << 30);  // type=2
  i32x8 g1 = {};
  g1[0] = (int)(1u << 16);                             // data_size = 2 bytes
  g1[1] = (int)((rowLen & 0xFFFFu) << 16);             // tensor_dim0[15:0]
  g1[2] = (int)(((rowLen >> 16) & 0xFFFFu) | ((rows & 0xFFFFu) << 16));
  g1[3] = (int)(((rows >> 16) & 0xFFFFu) | ((rowLen & 0xFFFFu) << 16)); // tile_dim0
  g1[4] = (int)(rows & 0xFFFFu);                       // tile_dim1, tile_dim2=0
  g1[5] = (int)strideElems;                            // tensor_dim0_stride[31:0]
  g1[6] = 0;
  g1[7] = 0;
  i32x4 gz = {};
#if defined(__clang_major__) && __clang_major__ >= 23
  i32x8 z8 = {};
  __builtin_amdgcn_tensor_load_to_lds(g0, g1, gz, gz, z8, 0);
#else
  __builtin_amdgcn_tensor_load_to_lds(g0, g1, gz, gz, 0);
#endif
}
__device__ __forceinline__ unsigned lds_off(const void* p) {
  return (unsigned)(uintptr_t)p;   // LDS aperture: addr[31:0] = LDS byte offset
}
#endif

// ---- f32 -> f16 conversion kernels ----------------------------------------
__global__ void cvt_kernel(const float* __restrict__ src,
                           _Float16* __restrict__ dst, int n) {
  int i = blockIdx.x * blockDim.x + threadIdx.x;
  if (i < n) dst[i] = (_Float16)src[i];
}

// dst[n*K + k] = src[k*N + n]  (transpose to N-major / K-contiguous)
__global__ void cvtT_kernel(const float* __restrict__ src,
                            _Float16* __restrict__ dst, int K, int N) {
  int i = blockIdx.x * blockDim.x + threadIdx.x;
  if (i < K * N) {
    int k = i / N, n = i % N;
    dst[(size_t)n * K + k] = (_Float16)src[i];
  }
}

// ---- Generic WMMA GEMM: C[M,N] = A[M,K] * B[K,N] + bias -------------------
// A: row-major f16. Bt: B stored transposed (Bt[n*K+k] = B[k,n]), f16.
// 128 threads = 4 waves; each wave computes a 16x64 strip; block tile 64x64.
// K-loop is software-pipelined AND 2x unrolled: the two fragment register
// sets alternate by name, so stage n+1's 10 loads issue before stage n's
// WMMAs with no rotation copies; s_wait_loadcnt drains incrementally.
__global__ __launch_bounds__(128)
void gemm_wmma_kernel(const _Float16* __restrict__ A,
                      const _Float16* __restrict__ Bt,
                      const float* __restrict__ bias,
                      _Float16* __restrict__ out16,
                      float* __restrict__ out32,
                      int M, int N, int K, int transposedOut) {
  const int w    = threadIdx.x >> 5;
  const int lane = threadIdx.x & 31;
  const int m0   = blockIdx.y * 64 + w * 16;
  const int n0   = blockIdx.x * 64;

  const _Float16* Arow = A + (size_t)m0 * K;
  const _Float16* Bn[4];
#pragma unroll
  for (int j = 0; j < 4; ++j) Bn[j] = Bt + (size_t)(n0 + j * 16) * K;

  v8f acc[4] = {};
  v16h a0 = load_frag_a(Arow, K);
  v16h b0[4];
#pragma unroll
  for (int j = 0; j < 4; ++j) b0[j] = load_frag_b(Bn[j], K);

  int k = 0;
  for (; k + 64 <= K; k += 64) {
    // stage 1 loads (k+32) issue before stage 0 WMMAs consume a0/b0
    v16h a1 = load_frag_a(Arow + k + 32, K);
    v16h b1[4];
#pragma unroll
    for (int j = 0; j < 4; ++j) b1[j] = load_frag_b(Bn[j] + k + 32, K);
#pragma unroll
    for (int j = 0; j < 4; ++j) acc[j] = wmma_f16(a0, b0[j], acc[j]);

    if (k + 64 < K) {       // stage 0 reload (k+64) before stage 1 WMMAs
      __builtin_prefetch(Arow + k + 96, 0, 3);
      a0 = load_frag_a(Arow + k + 64, K);
#pragma unroll
      for (int j = 0; j < 4; ++j) b0[j] = load_frag_b(Bn[j] + k + 64, K);
    }
#pragma unroll
    for (int j = 0; j < 4; ++j) acc[j] = wmma_f16(a1, b1[j], acc[j]);
  }
  if (k < K) {              // tail when K % 64 == 32 (unused for K=768)
#pragma unroll
    for (int j = 0; j < 4; ++j) acc[j] = wmma_f16(a0, b0[j], acc[j]);
  }

  const int mh = (lane & 16) ? 8 : 0;
  const int nl = lane & 15;
#pragma unroll
  for (int j = 0; j < 4; ++j) {
    const int n = n0 + j * 16 + nl;
    const float bv = bias ? bias[n] : 0.0f;
#pragma unroll
    for (int i = 0; i < 8; ++i) {
      const int m = m0 + mh + i;
      const float v = acc[j][i] + bv;
      if (out16) {
        if (transposedOut) out16[(size_t)n * M + m] = (_Float16)v;
        else               out16[(size_t)m * N + n] = (_Float16)v;
      }
      if (out32) out32[(size_t)m * N + n] = v;
    }
  }
}

// ---- Block-sparse attention (one workgroup per (head, q-block)) -----------
// Q16,K16: (T, D) f16, head h occupies cols [h*64, h*64+64).
// VT16: per-column transposed: VT16[(h*64+d)*T + t] = V[t, h*64+d].
// K/V 64x64 tiles are staged into double-buffered LDS via the TDM and
// consumed through ds_load fragment reads; P round-trips LDS to convert the
// D-fragment layout into an A-fragment layout for the P*V WMMA.
__global__ __launch_bounds__(128)
void attn_kernel(const _Float16* __restrict__ Q16,
                 const _Float16* __restrict__ K16,
                 const _Float16* __restrict__ VT16,
                 const int* __restrict__ mask,       // (T,)
                 const int* __restrict__ rb,         // (NB-2, 3)
                 _Float16* __restrict__ ctx16) {
  const int h    = blockIdx.x / NBdim;
  const int g    = blockIdx.x % NBdim;
  const int w    = threadIdx.x >> 5;
  const int lane = threadIdx.x & 31;

  __shared__ __align__(32) _Float16 Klds[2][64 * 64];  // [key][dim]
  __shared__ __align__(32) _Float16 Vlds[2][64 * 64];  // [dim][key]
  __shared__ __align__(32) _Float16 Plds[4][16 * 64];

  int kbl[NBdim];
  int nkb;
  if (g == 0 || g == NBdim - 1) {       // global blocks attend to all keys
    nkb = NBdim;
#pragma unroll
    for (int i = 0; i < NBdim; ++i) kbl[i] = i;
  } else {                              // 2 global + 3 window + 3 random
    nkb = 8;
    kbl[0] = 0;  kbl[1] = NBdim - 1;
    kbl[2] = g - 1; kbl[3] = g; kbl[4] = g + 1;
    kbl[5] = rb[(g - 1) * 3 + 0];
    kbl[6] = rb[(g - 1) * 3 + 1];
    kbl[7] = rb[(g - 1) * 3 + 2];
  }

  // Q strip for this wave: rows [g*64 + w*16, +16), dims [h*64, +64)
  const _Float16* qbase = Q16 + (size_t)(g * 64 + w * 16) * Ddim + h * HDdim;
  v16h aq0 = load_frag_a(qbase, Ddim);
  v16h aq1 = load_frag_a(qbase + 32, Ddim);

  v8f o[4] = {};
  float rm[8], rs[8];
#pragma unroll
  for (int i = 0; i < 8; ++i) { rm[i] = -3.0e38f; rs[i] = 0.0f; }

  const int mh = (lane & 16) ? 8 : 0;
  const int nl = lane & 15;

#if HAVE_TDM
  if (w == 0) {  // TDM ops are wave-level (EXEC ignored); wave 0 issues
    const int kt = kbl[0] * 64;
    tdm_load_tile_2d(lds_off(Klds[0]), K16 + (size_t)kt * Ddim + h * HDdim,
                     64, 64, Ddim);
    tdm_load_tile_2d(lds_off(Vlds[0]), VT16 + (size_t)(h * HDdim) * Tdim + kt,
                     64, 64, Tdim);
  }
#endif

  for (int it = 0; it < nkb; ++it) {
    const int kt0 = kbl[it] * 64;
    const int buf = it & 1;

#if HAVE_TDM
    if (w == 0) {
      if (it + 1 < nkb) {     // kick off next tile pair into the other buffer
        const int ktn = kbl[it + 1] * 64;
        tdm_load_tile_2d(lds_off(Klds[buf ^ 1]),
                         K16 + (size_t)ktn * Ddim + h * HDdim, 64, 64, Ddim);
        tdm_load_tile_2d(lds_off(Vlds[buf ^ 1]),
                         VT16 + (size_t)(h * HDdim) * Tdim + ktn, 64, 64, Tdim);
        __builtin_amdgcn_s_wait_tensorcnt(2);  // current pair complete
      } else {
        __builtin_amdgcn_s_wait_tensorcnt(0);
      }
    }
    __syncthreads();          // staged tiles visible to all waves
#else
    __syncthreads();          // WAR: previous reads done before overwrite
    for (int c = threadIdx.x; c < 512; c += 128) {  // 8KB per tile, b128 copies
      const int r = c >> 3, off = (c & 7) * 8;
      *(v8h*)&Klds[buf][r * 64 + off] =
          *(const v8h*)&K16[(size_t)(kt0 + r) * Ddim + h * HDdim + off];
      *(v8h*)&Vlds[buf][r * 64 + off] =
          *(const v8h*)&VT16[(size_t)(h * HDdim + r) * Tdim + kt0 + off];
    }
    __syncthreads();
#endif

    // S = Q * K^T  (16x64 scores in 4 fragments), K from LDS tile
    v8f s[4] = {};
#pragma unroll
    for (int j = 0; j < 4; ++j) {
      const _Float16* kp = &Klds[buf][(j * 16) * 64];
      s[j] = wmma_f16(aq0, load_frag_b(kp, 64), s[j]);
      s[j] = wmma_f16(aq1, load_frag_b(kp + 32, 64), s[j]);
    }

    // scale + mask (column-wise token mask)
#pragma unroll
    for (int j = 0; j < 4; ++j) {
      const int kt = kt0 + j * 16 + nl;
      const bool ok = mask[kt] > 0;
#pragma unroll
      for (int i = 0; i < 8; ++i)
        s[j][i] = ok ? s[j][i] * SCALE : NEGV;
    }

    // online softmax: block row-max (reduce over 4 frags + 16 lanes of N)
    float bm[8];
#pragma unroll
    for (int i = 0; i < 8; ++i)
      bm[i] = fmaxf(fmaxf(s[0][i], s[1][i]), fmaxf(s[2][i], s[3][i]));
#pragma unroll
    for (int off = 8; off >= 1; off >>= 1)
#pragma unroll
      for (int i = 0; i < 8; ++i)
        bm[i] = fmaxf(bm[i], __shfl_xor(bm[i], off, 32));

    float corr[8];
#pragma unroll
    for (int i = 0; i < 8; ++i) {
      float nm2 = fmaxf(rm[i], bm[i]);
      corr[i] = __expf(rm[i] - nm2);
      rm[i] = nm2;
    }

    float ps[8];
#pragma unroll
    for (int i = 0; i < 8; ++i) ps[i] = 0.0f;
#pragma unroll
    for (int j = 0; j < 4; ++j)
#pragma unroll
      for (int i = 0; i < 8; ++i) {
        float pv = __expf(s[j][i] - rm[i]);
        s[j][i] = pv;
        ps[i] += pv;
      }
#pragma unroll
    for (int off = 8; off >= 1; off >>= 1)
#pragma unroll
      for (int i = 0; i < 8; ++i) ps[i] += __shfl_xor(ps[i], off, 32);
#pragma unroll
    for (int i = 0; i < 8; ++i) rs[i] = rs[i] * corr[i] + ps[i];

    // rescale running context
#pragma unroll
    for (int j = 0; j < 4; ++j)
#pragma unroll
      for (int i = 0; i < 8; ++i) o[j][i] *= corr[i];

    // Stage P (D-fragment layout) into per-wave LDS, reload as A fragments.
    // LDS ops are in-order within a wave; only a compiler barrier needed.
    _Float16* P = Plds[w];
#pragma unroll
    for (int j = 0; j < 4; ++j)
#pragma unroll
      for (int i = 0; i < 8; ++i)
        P[(mh + i) * 64 + j * 16 + nl] = (_Float16)s[j][i];
    asm volatile("" ::: "memory");

    v16h ap0 = load_frag_a(P, 64);
    v16h ap1 = load_frag_a(P + 32, 64);

    // O += P * V  : B(k,n) = V[key kt0+k, h*64+n] from LDS tile [dim][key]
#pragma unroll
    for (int j = 0; j < 4; ++j) {
      const _Float16* vb = &Vlds[buf][(j * 16) * 64];
      o[j] = wmma_f16(ap0, load_frag_b(vb, 64), o[j]);
      o[j] = wmma_f16(ap1, load_frag_b(vb + 32, 64), o[j]);
    }
    __syncthreads();          // tile reads done before next TDM overwrite
  }

  // finalize: divide by softmax denominator, scatter to (T, D) f16
  float inv[8];
#pragma unroll
  for (int i = 0; i < 8; ++i) inv[i] = 1.0f / rs[i];
#pragma unroll
  for (int j = 0; j < 4; ++j)
#pragma unroll
    for (int i = 0; i < 8; ++i) {
      const int t = g * 64 + w * 16 + mh + i;
      ctx16[(size_t)t * Ddim + h * HDdim + j * 16 + nl] =
          (_Float16)(o[j][i] * inv[i]);
    }
}

// ---- Row LayerNorm: f32 in -> f16 out -------------------------------------
__global__ __launch_bounds__(256)
void ln_kernel(const float* __restrict__ hs, const float* __restrict__ gam,
               const float* __restrict__ bet, _Float16* __restrict__ out) {
  const int t = blockIdx.x;
  const float* row = hs + (size_t)t * Ddim;
  float s = 0.0f, ss = 0.0f;
  for (int i = threadIdx.x; i < Ddim; i += 256) {
    float v = row[i];
    s += v;
    ss += v * v;
  }
  __shared__ float red[2][8];
#pragma unroll
  for (int off = 16; off >= 1; off >>= 1) {
    s  += __shfl_xor(s, off, 32);
    ss += __shfl_xor(ss, off, 32);
  }
  if ((threadIdx.x & 31) == 0) {
    red[0][threadIdx.x >> 5] = s;
    red[1][threadIdx.x >> 5] = ss;
  }
  __syncthreads();
  s = 0.0f; ss = 0.0f;
#pragma unroll
  for (int i = 0; i < 8; ++i) { s += red[0][i]; ss += red[1][i]; }
  const float mu   = s / Ddim;
  const float var  = ss / Ddim - mu * mu;
  const float rstd = rsqrtf(var + 1e-12f);
  for (int i = threadIdx.x; i < Ddim; i += 256)
    out[(size_t)t * Ddim + i] = (_Float16)((row[i] - mu) * rstd * gam[i] + bet[i]);
}

// ---- Classifier head: (1024x768) x (768x16) via one wave per 16-row tile --
__global__ __launch_bounds__(32)
void head_kernel(const _Float16* __restrict__ A,     // (T, D) f16
                 const _Float16* __restrict__ WcT,   // WcT[c*D + k]
                 const float* __restrict__ bc, float* __restrict__ out) {
  const int m0   = blockIdx.x * 16;
  const int lane = threadIdx.x & 31;
  v8f acc = {};
  for (int k = 0; k < Ddim; k += 32)
    acc = wmma_f16(load_frag_a(A + (size_t)m0 * Ddim + k, Ddim),
                   load_frag_b(WcT + k, Ddim), acc);
  const int mh = (lane & 16) ? 8 : 0;
  const int n  = lane & 15;
#pragma unroll
  for (int i = 0; i < 8; ++i)
    out[(size_t)(m0 + mh + i) * Cdim + n] = acc[i] + bc[n];
}

// ---------------------------------------------------------------------------
extern "C" void kernel_launch(void* const* d_in, const int* in_sizes, int n_in,
                              void* d_out, int out_size, void* d_ws, size_t ws_size,
                              hipStream_t stream) {
  const float* hidden = (const float*)d_in[0];
  const int*   amask  = (const int*)d_in[1];
  const int*   rblk   = (const int*)d_in[2];
  const float* Wq = (const float*)d_in[3];   const float* bq = (const float*)d_in[4];
  const float* Wk = (const float*)d_in[5];   const float* bk = (const float*)d_in[6];
  const float* Wv = (const float*)d_in[7];   const float* bv = (const float*)d_in[8];
  const float* Wo = (const float*)d_in[9];   const float* bo = (const float*)d_in[10];
  const float* Wd = (const float*)d_in[11];  const float* bd = (const float*)d_in[12];
  const float* lg = (const float*)d_in[13];  const float* lb = (const float*)d_in[14];
  const float* Wc = (const float*)d_in[15];  const float* bc = (const float*)d_in[16];

  const size_t TD = (size_t)Tdim * Ddim;   // 786432
  const size_t DD = (size_t)Ddim * Ddim;   // 589824

  _Float16* p = (_Float16*)d_ws;
  _Float16* h16   = p; p += TD;
  _Float16* WqT   = p; p += DD;
  _Float16* WkT   = p; p += DD;
  _Float16* WvT   = p; p += DD;
  _Float16* WoT   = p; p += DD;
  _Float16* WdT   = p; p += DD;
  _Float16* WcT   = p; p += (size_t)Ddim * Cdim;
  _Float16* Q16   = p; p += TD;
  _Float16* K16   = p; p += TD;
  _Float16* VT16  = p; p += TD;
  _Float16* ctx16 = p; p += TD;
  _Float16* ao16  = p; p += TD;
  _Float16* hsln  = p; p += TD;
  float*    hs32  = (float*)p;   // all previous counts keep 4B alignment

  // Only the last layer's hidden states reach the output.
  cvt_kernel<<<(int)((TD + 255) / 256), 256, 0, stream>>>(
      hidden + (size_t)(Ldim - 1) * TD, h16, (int)TD);
  cvtT_kernel<<<(int)((DD + 255) / 256), 256, 0, stream>>>(Wq, WqT, Ddim, Ddim);
  cvtT_kernel<<<(int)((DD + 255) / 256), 256, 0, stream>>>(Wk, WkT, Ddim, Ddim);
  cvtT_kernel<<<(int)((DD + 255) / 256), 256, 0, stream>>>(Wv, WvT, Ddim, Ddim);
  cvtT_kernel<<<(int)((DD + 255) / 256), 256, 0, stream>>>(Wo, WoT, Ddim, Ddim);
  cvtT_kernel<<<(int)((DD + 255) / 256), 256, 0, stream>>>(Wd, WdT, Ddim, Ddim);
  cvtT_kernel<<<(Ddim * Cdim + 255) / 256, 256, 0, stream>>>(Wc, WcT, Ddim, Cdim);

  dim3 gTile(Ddim / 64, Tdim / 64);  // (12, 16)
  // QKV projections (V stored per-column transposed for the P*V WMMA)
  gemm_wmma_kernel<<<gTile, 128, 0, stream>>>(h16, WqT, bq, Q16, nullptr,
                                              Tdim, Ddim, Ddim, 0);
  gemm_wmma_kernel<<<gTile, 128, 0, stream>>>(h16, WkT, bk, K16, nullptr,
                                              Tdim, Ddim, Ddim, 0);
  gemm_wmma_kernel<<<gTile, 128, 0, stream>>>(h16, WvT, bv, VT16, nullptr,
                                              Tdim, Ddim, Ddim, 1);

  // Block-sparse attention: one workgroup per (head, q-block)
  attn_kernel<<<Hdim * NBdim, 128, 0, stream>>>(Q16, K16, VT16, amask, rblk,
                                                ctx16);

  // Output projection + dense
  gemm_wmma_kernel<<<gTile, 128, 0, stream>>>(ctx16, WoT, bo, ao16, nullptr,
                                              Tdim, Ddim, Ddim, 0);
  gemm_wmma_kernel<<<gTile, 128, 0, stream>>>(ao16, WdT, bd, nullptr, hs32,
                                              Tdim, Ddim, Ddim, 0);

  // LayerNorm (f32 stats), then classifier head
  ln_kernel<<<Tdim, 256, 0, stream>>>(hs32, lg, lb, hsln);
  head_kernel<<<Tdim / 16, 32, 0, stream>>>(hsln, WcT, bc, (float*)d_out);
}
